// BidirectionalRNN_20255065768520
// MI455X (gfx1250) — matrix operand order repaired and checked by reading, not compiled, not run on hardware
//
#include <hip/hip_runtime.h>
#include <hip/hip_bf16.h>

// ---------------------------------------------------------------------------
// Bidirectional ragged GRU for MI455X (gfx1250, wave32, WMMA bf16 path).
//   Phase 1: xw = xb @ W_ih + b_ih  (ragged reverse folded into gather,
//            W_ih fragments register-resident across 8 time tiles/block)
//   Phase 2: persistent-workgroup recurrent scan, W_hh resident in LDS as
//            bf16 B-fragments, h ping-ponged in LDS as bf16 A-fragments,
//            2-step-deep register pipeline + global_prefetch_b8 for xw.
// Scratch: d_ws holds xw = 64*2048*384 f32 = 201,326,592 bytes.
// ---------------------------------------------------------------------------

typedef __attribute__((ext_vector_type(16))) __bf16 v16bf;
typedef __attribute__((ext_vector_type(8)))  float  v8f;

#define NB 32      // sequences per direction
#define BB 64      // total batch rows (fwd + reversed)
#define LL 2048    // time steps
#define CC 64      // input channels
#define HH 128     // hidden size
#define GG 384     // 3*HH gate width
#define TPB 8      // time tiles per block in kernel 1
#define PFD 12     // deep-prefetch distance (steps) in kernel 2

#if __has_builtin(__builtin_amdgcn_rcpf)
static __device__ __forceinline__ float fast_rcp(float x) { return __builtin_amdgcn_rcpf(x); }
#else
static __device__ __forceinline__ float fast_rcp(float x) { return 1.0f / x; }
#endif

#if __has_builtin(__builtin_amdgcn_tanhf)
static __device__ __forceinline__ float fast_tanh(float x) { return __builtin_amdgcn_tanhf(x); }
#else
static __device__ __forceinline__ float fast_tanh(float x) { return tanhf(x); }
#endif

static __device__ __forceinline__ float sigmoidf_(float x) {
    return fast_rcp(1.0f + __expf(-x));
}

static __device__ __forceinline__ v8f wmma_bf16(v16bf a, v16bf b, v8f c) {
    // (neg_a, A, neg_b, B, c_mod, C, reuse_a, reuse_b)
    return __builtin_amdgcn_wmma_f32_16x16x32_bf16(false, a, false, b,
                                                   (short)0, c, false, false);
}

static __device__ __forceinline__ void pack8(__bf16* dst, float4 a, float4 b) {
    dst[0] = (__bf16)a.x; dst[1] = (__bf16)a.y;
    dst[2] = (__bf16)a.z; dst[3] = (__bf16)a.w;
    dst[4] = (__bf16)b.x; dst[5] = (__bf16)b.y;
    dst[6] = (__bf16)b.z; dst[7] = (__bf16)b.w;
}

// ===========================================================================
// Kernel 1: xw[b][t][0:384] = xb[b][t][:] @ W_ih + b_ih
//   grid = (LL/(16*TPB), BB), block = 256 (8 waves). Wave w covers gate
//   columns {16w, 16w+128, 16w+256}. W_ih fragments live in VGPRs across
//   TPB time tiles; x tile gathered with float4 loads (ragged reverse here).
// ===========================================================================
__global__ __launch_bounds__(256) void k_input_gemm(
    const float* __restrict__ x, const int* __restrict__ lengths,
    const float* __restrict__ Wih, const float* __restrict__ bih,
    float* __restrict__ xw)
{
    const int tid = threadIdx.x;
    const int w  = tid >> 5;
    const int l  = tid & 31;
    const int lm = l & 15;
    const int lh = l >> 4;
    const int b  = blockIdx.y;
    const int tbase = blockIdx.x * (16 * TPB);

    // ---- one-time: W_ih bf16 B-fragments + biases, register-resident ----
    v16bf bf[3][2];
    float bi[3];
    #pragma unroll
    for (int g = 0; g < 3; ++g) {
        const int col = 128 * g + 16 * w + lm;
        #pragma unroll
        for (int s = 0; s < 2; ++s) {
            #pragma unroll
            for (int i = 0; i < 16; ++i) {
                const int k = 32 * s + 8 * lh + i + (i >= 8 ? 8 : 0);
                ((__bf16*)&bf[g][s])[i] = (__bf16)Wih[(size_t)k * GG + col];
            }
        }
        bi[g] = bih[col];
    }

    int len = 0;
    if (b >= NB) len = lengths[b - NB];

    for (int tt0 = 0; tt0 < TPB; ++tt0) {
        const int t0 = tbase + tt0 * 16;
        const int tt = t0 + lm;              // A layout: M = time row = lm

        // ---- A fragments (16x32 bf16): ragged-reverse gather, float4 ----
        const float* xrow = nullptr;
        if (b < NB) {
            xrow = x + ((size_t)b * LL + tt) * CC;
        } else if (tt < len) {
            xrow = x + ((size_t)(b - NB) * LL + (len - 1 - tt)) * CC;
        }
        v16bf a[2];
        if (xrow) {
            #pragma unroll
            for (int s = 0; s < 2; ++s) {
                const float* p = xrow + 32 * s + 8 * lh;
                float4 q0 = *(const float4*)(p);
                float4 q1 = *(const float4*)(p + 4);
                float4 q2 = *(const float4*)(p + 16);
                float4 q3 = *(const float4*)(p + 20);
                pack8(((__bf16*)&a[s]),     q0, q1);
                pack8(((__bf16*)&a[s]) + 8, q2, q3);
            }
        } else {
            #pragma unroll
            for (int s = 0; s < 2; ++s)
                #pragma unroll
                for (int i = 0; i < 16; ++i)
                    ((__bf16*)&a[s])[i] = (__bf16)0.0f;
        }

        // ---- 3 gate tiles: 2 WMMAs each, store D (coalesced across lanes) --
        #pragma unroll
        for (int g = 0; g < 3; ++g) {
            const float s0 = bi[g];
            v8f c = {s0, s0, s0, s0, s0, s0, s0, s0};
            c = wmma_bf16(a[0], bf[g][0], c);
            c = wmma_bf16(a[1], bf[g][1], c);

            float* orow = xw + ((size_t)b * LL + t0) * GG + 128 * g + 16 * w + lm;
            #pragma unroll
            for (int v = 0; v < 8; ++v) {
                const int m = v + 8 * lh;    // D layout: M = v + 8*(l/16)
                orow[(size_t)m * GG] = c[v];
            }
        }
    }
}

// ===========================================================================
// Kernel 2: recurrent scan. grid = 4 workgroups x 16 batch rows, 8 waves.
//   Wave w owns hidden columns j in [16w, 16w+16) for all three gates.
//   Per step/wave: 12x v_wmma_f32_16x16x32_bf16 (3 gates x K=128/32).
//   xw pipelined 2 steps deep in registers + global_prefetch_b8 PFD ahead.
// ===========================================================================
__global__ __launch_bounds__(256) void k_gru_scan(
    const float* __restrict__ xw, const float* __restrict__ Whh,
    const float* __restrict__ bhh, float* __restrict__ out)
{
    __shared__ v16bf sB[8][3][4][32];   // W_hh bf16 B-fragments: 96 KB
    __shared__ v16bf sA[2][4][32];      // h bf16 A-fragments (ping-pong): 8 KB

    const int tid = threadIdx.x;
    const int w   = tid >> 5;
    const int l   = tid & 31;
    const int lm  = l & 15;
    const int lh  = l >> 4;
    const int bb  = blockIdx.x * 16;    // batch tile base
    const int j0  = 16 * w + lm;        // this lane's hidden column

    // ---- one-time: W_hh -> bf16 B-fragments in LDS (per-lane slots) ----
    #pragma unroll
    for (int g = 0; g < 3; ++g) {
        #pragma unroll
        for (int s = 0; s < 4; ++s) {
            v16bf f;
            #pragma unroll
            for (int i = 0; i < 16; ++i) {
                const int k = 32 * s + 8 * lh + i + (i >= 8 ? 8 : 0);
                ((__bf16*)&f)[i] = (__bf16)Whh[(size_t)k * GG + 128 * g + j0];
            }
            sB[w][g][s][l] = f;
        }
    }

    // ---- h0 = 0: zero ping buffer 0 ----
    if (w == 0) {
        v16bf z;
        #pragma unroll
        for (int i = 0; i < 16; ++i) ((__bf16*)&z)[i] = (__bf16)0.0f;
        #pragma unroll
        for (int s = 0; s < 4; ++s) sA[0][s][l] = z;
    }

    const float bhr = bhh[j0];
    const float bhz = bhh[128 + j0];
    const float bhn = bhh[256 + j0];

    // per-v (row m = v + 8*lh) global base pointers
    const float* pxw[8];
    float*       pout[8];
    #pragma unroll
    for (int v = 0; v < 8; ++v) {
        const int b = bb + v + 8 * lh;
        pxw[v] = xw + (size_t)b * LL * GG + j0;
        const int bo = (b < NB) ? b : (b - NB);
        const int co = (b < NB) ? 0 : HH;   // fwd -> cols [0,128), bwd -> [128,256)
        pout[v] = out + (size_t)bo * LL * (2 * HH) + co + j0;
    }

    float hold[8];
    #pragma unroll
    for (int v = 0; v < 8; ++v) hold[v] = 0.0f;

    // scatter constants for writing h_new into A-fragment layout
    const int s_w = j0 >> 5;
    const int ko  = j0 & 31;
    const int i_w = (ko & 7) + ((ko >> 4) & 1) * 8;
    const int b3  = (ko >> 3) & 1;

    __syncthreads();

    // xc layout: [0..7]=r, [8..15]=z, [16..23]=n
    auto loadx = [&](int t, float* xc) {
        #pragma unroll
        for (int v = 0; v < 8; ++v) {
            const float* p = pxw[v] + (size_t)t * GG;
            xc[v]      = p[0];
            xc[8 + v]  = p[128];
            xc[16 + v] = p[256];
        }
    };

    auto step = [&](int t, float* xc) {
        const int rb = t & 1;
        const int wbuf = rb ^ 1;

        // load h_t A-fragments (contiguous 32B per lane)
        v16bf a0 = sA[rb][0][l];
        v16bf a1 = sA[rb][1][l];
        v16bf a2 = sA[rb][2][l];
        v16bf a3 = sA[rb][3][l];

        // gh = h @ W_hh + b_hh, f32 accumulate
        v8f cr = {bhr, bhr, bhr, bhr, bhr, bhr, bhr, bhr};
        v8f cz = {bhz, bhz, bhz, bhz, bhz, bhz, bhz, bhz};
        v8f cn = {bhn, bhn, bhn, bhn, bhn, bhn, bhn, bhn};

        cr = wmma_bf16(a0, sB[w][0][0][l], cr);
        cz = wmma_bf16(a0, sB[w][1][0][l], cz);
        cn = wmma_bf16(a0, sB[w][2][0][l], cn);
        cr = wmma_bf16(a1, sB[w][0][1][l], cr);
        cz = wmma_bf16(a1, sB[w][1][1][l], cz);
        cn = wmma_bf16(a1, sB[w][2][1][l], cn);
        cr = wmma_bf16(a2, sB[w][0][2][l], cr);
        cz = wmma_bf16(a2, sB[w][1][2][l], cz);
        cn = wmma_bf16(a2, sB[w][2][2][l], cn);
        cr = wmma_bf16(a3, sB[w][0][3][l], cr);
        cz = wmma_bf16(a3, sB[w][1][3][l], cz);
        cn = wmma_bf16(a3, sB[w][2][3][l], cn);

        // gate math + h update; scatter h_new into next A buffer
        #pragma unroll
        for (int v = 0; v < 8; ++v) {
            const float r  = sigmoidf_(xc[v]      + cr[v]);
            const float z  = sigmoidf_(xc[8 + v]  + cz[v]);
            const float n  = fast_tanh(xc[16 + v] + r * cn[v]);
            const float hn = (1.0f - z) * n + z * hold[v];
            hold[v] = hn;

            const int m = v + 8 * lh;                        // batch row in tile
            ((__bf16*)&sA[wbuf][s_w][m + 16 * b3])[i_w] = (__bf16)hn;
            pout[v][(size_t)t * (2 * HH)] = hn;              // fire-and-forget
        }

        // refill this buffer for step t+2 (wait lands two steps from now)
        if (t + 2 < LL) loadx(t + 2, xc);

        // deep prefetch into cache, PFD steps ahead (2 lanes per wave issue)
        if ((t + PFD < LL) && (lm == 0)) {
            #pragma unroll
            for (int v = 0; v < 8; ++v) {
                const float* p = pxw[v] + (size_t)(t + PFD) * GG;
                __builtin_prefetch(p,       0, 1);
                __builtin_prefetch(p + 128, 0, 1);
                __builtin_prefetch(p + 256, 0, 1);
            }
        }

        __syncthreads();   // publish h_{t+1} fragments to all 8 waves
    };

    float xa[24], xb[24];
    loadx(0, xa);
    loadx(1, xb);

    for (int t = 0; t < LL; t += 2) {
        step(t,     xa);
        step(t + 1, xb);
    }
}

// ===========================================================================
extern "C" void kernel_launch(void* const* d_in, const int* in_sizes, int n_in,
                              void* d_out, int out_size, void* d_ws, size_t ws_size,
                              hipStream_t stream) {
    const float* x       = (const float*)d_in[0];
    const int*   lengths = (const int*)  d_in[1];
    const float* Wih     = (const float*)d_in[2];
    const float* Whh     = (const float*)d_in[3];
    const float* bih     = (const float*)d_in[4];
    const float* bhh     = (const float*)d_in[5];
    float*       out     = (float*)d_out;
    float*       xwbuf   = (float*)d_ws;   // needs BB*LL*GG*4 = 201,326,592 B

    dim3 g1(LL / (16 * TPB), BB);
    k_input_gemm<<<g1, 256, 0, stream>>>(x, lengths, Wih, bih, xwbuf);
    k_gru_scan<<<4, 256, 0, stream>>>(xwbuf, Whh, bhh, out);
}
